// NodeEdgeGATEncoder_15504831939064
// MI455X (gfx1250) — compile-verified, matrix operand order checked by reading, not run on hardware
//
#include <hip/hip_runtime.h>
#include <hip/hip_bf16.h>
#include <math.h>

// ---------------------------------------------------------------------------
// CDNA5 (gfx1250) WMMA types
// ---------------------------------------------------------------------------
typedef __attribute__((ext_vector_type(16))) __bf16 v16bf;
typedef __attribute__((ext_vector_type(8)))  float  v8f;

#define HID 64
#define NEG_SLOPE 0.2f

static __device__ __forceinline__ v8f wmma_bf16(v16bf a, v16bf b, v8f c) {
    return __builtin_amdgcn_wmma_f32_16x16x32_bf16(
        /*neg_a=*/false, a, /*neg_b=*/false, b,
        /*c_mod=*/(short)0, c, /*reuse_a=*/false, /*reuse_b=*/false);
}

// ---------------------------------------------------------------------------
// A fragment loader: 16x32 (MxK) tile, fp32 row-major, ld = row stride.
// Partial tiles: CLAMP the row (duplicated rows are never stored), so all
// loads are unconditional 16B vector loads.
// ---------------------------------------------------------------------------
static __device__ __forceinline__ v16bf load_A_f32(const float* __restrict__ src,
                                                   int ld, long long row0,
                                                   int k_base, int lane,
                                                   long long M) {
    long long r = row0 + (lane & 15);
    if (r >= M) r = M - 1;
    int kh = (lane >> 4) * 8;
    const float* p = src + r * (long long)ld + k_base + kh;
    float4 a0 = *(const float4*)(p + 0);
    float4 a1 = *(const float4*)(p + 4);
    float4 b0 = *(const float4*)(p + 16);
    float4 b1 = *(const float4*)(p + 20);
    v16bf v;
    v[0]  = (__bf16)a0.x; v[1]  = (__bf16)a0.y; v[2]  = (__bf16)a0.z; v[3]  = (__bf16)a0.w;
    v[4]  = (__bf16)a1.x; v[5]  = (__bf16)a1.y; v[6]  = (__bf16)a1.z; v[7]  = (__bf16)a1.w;
    v[8]  = (__bf16)b0.x; v[9]  = (__bf16)b0.y; v[10] = (__bf16)b0.z; v[11] = (__bf16)b0.w;
    v[12] = (__bf16)b1.x; v[13] = (__bf16)b1.y; v[14] = (__bf16)b1.z; v[15] = (__bf16)b1.w;
    return v;
}

// B fragment from LDS-staged pack: 32 contiguous bytes per lane.
static __device__ __forceinline__ v16bf frag_B(const __bf16* Bs, int kt, int nt,
                                               int lane) {
    return *(const v16bf*)(Bs + (((kt * 4 + nt) * 32 + lane) << 4));
}

// D store: lane<16 -> N=lane, rows M=r ; lane>=16 -> N=lane-16, rows M=8+r
static __device__ __forceinline__ void store_D(float* __restrict__ dst, int ld,
                                               long long row0, int col_base,
                                               v8f d, int lane, long long M) {
    int n  = col_base + (lane & 15);
    int mb = (lane < 16) ? 0 : 8;
    #pragma unroll
    for (int r = 0; r < 8; ++r) {
        long long m = row0 + mb + r;
        if (m < M) dst[m * (long long)ld + n] = d[r];
    }
}

static __device__ __forceinline__ void atomicMaxF(float* addr, float val) {
    if (val >= 0.f) atomicMax((int*)addr, __float_as_int(val));
    else            atomicMin((unsigned int*)addr, __float_as_uint(val));
}

// ---------------------------------------------------------------------------
// Weight packer: [K x 64] fp32 row-major -> bf16 B-fragment layout:
// element index t = ((kt*4+nt)*32 + lane)*16 + i
// ---------------------------------------------------------------------------
__global__ void pack_weight(const float* __restrict__ W, __bf16* __restrict__ out,
                            int K) {
    int t = blockIdx.x * blockDim.x + threadIdx.x;
    if (t >= K * HID) return;
    int i    = t & 15;
    int lane = (t >> 4) & 31;
    int f    = t >> 9;            // kt*4 + nt
    int kt = f >> 2, nt = f & 3;
    int n  = nt * 16 + (lane & 15);
    int kh = (lane >> 4) * 8;
    int k  = kt * 32 + ((i < 8) ? (kh + i) : (16 + kh + (i - 8)));
    out[t] = (__bf16)W[k * HID + n];
}

// ---------------------------------------------------------------------------
// Utility kernels
// ---------------------------------------------------------------------------
__global__ void fill_f32(float* __restrict__ p, float v, long long n) {
    long long t = (long long)blockIdx.x * blockDim.x + threadIdx.x;
    if (t < n) p[t] = v;
}

__global__ void copy_i32(const int* __restrict__ in, int* __restrict__ out, int n) {
    int t = blockIdx.x * blockDim.x + threadIdx.x;
    if (t < n) out[t] = in[t];
}

__global__ void matvec64(const float* __restrict__ W, const float* __restrict__ a,
                         float* __restrict__ out) {
    int k = threadIdx.x;
    float s = 0.f;
    #pragma unroll
    for (int h = 0; h < HID; ++h) s += W[k * HID + h] * a[h];
    out[k] = s;
}

__global__ void edge_deg_loop(const int* __restrict__ dst,
                              const float* __restrict__ edge_attr,
                              float* __restrict__ deg, float* __restrict__ loop_attr,
                              int E) {
    long long t = (long long)blockIdx.x * blockDim.x + threadIdx.x;
    if (t >= (long long)E * 8) return;
    int e  = (int)(t >> 3);
    int c0 = (int)(t & 7) * 8;
    int d  = dst[e];
    if (c0 == 0) atomicAdd(&deg[d], 1.0f);
    const float* row = edge_attr + (long long)e * HID;
    float* acc = loop_attr + (long long)d * HID;
    #pragma unroll
    for (int c = 0; c < 8; ++c) atomicAdd(&acc[c0 + c], row[c0 + c]);
}

__global__ void finalize_loop(float* __restrict__ loop_attr,
                              const float* __restrict__ deg, int N) {
    long long t = (long long)blockIdx.x * blockDim.x + threadIdx.x;
    if (t >= (long long)N * HID) return;
    int i = (int)(t >> 6);
    float dg = deg[i];
    loop_attr[t] /= (dg > 1.f ? dg : 1.f);
}

// ---------------------------------------------------------------------------
// WMMA node projection: Y[M x 64] = X[M x K] @ W[K x 64].
// Packed bf16 B staged in LDS once per block (no VGPR spills); A loaded once
// per tile with float4 loads; grid-stride loop over 16-row tiles.
// ---------------------------------------------------------------------------
template <int KT>
__global__ __launch_bounds__(256) void gemm_proj_t(const float* __restrict__ X,
                                                   const __bf16* __restrict__ Bp,
                                                   float* __restrict__ Y,
                                                   long long M) {
    const int K = KT * 32;
    __shared__ __bf16 Bs[KT * 4 * 32 * 16];          // KT*4 KB
    for (int t = threadIdx.x; t < KT * 256; t += 256)
        ((int4*)Bs)[t] = ((const int4*)Bp)[t];
    __syncthreads();

    int lane = threadIdx.x & 31;
    int wave = threadIdx.x >> 5;
    long long ntiles = (M + 15) >> 4;
    long long stride = (long long)gridDim.x * 8;
    for (long long tile = (long long)blockIdx.x * 8 + wave; tile < ntiles;
         tile += stride) {
        long long row0 = tile << 4;
        v16bf A[KT];
        #pragma unroll
        for (int kt = 0; kt < KT; ++kt)
            A[kt] = load_A_f32(X, K, row0, kt * 32, lane, M);
        #pragma unroll
        for (int nt = 0; nt < 4; ++nt) {
            v8f acc = {};
            #pragma unroll
            for (int kt = 0; kt < KT; ++kt)
                acc = wmma_bf16(A[kt], frag_B(Bs, kt, nt, lane), acc);
            store_D(Y, HID, row0, nt * 16, acc, lane, M);
        }
    }
}

// ---------------------------------------------------------------------------
// Attention scalar / scatter kernels
// ---------------------------------------------------------------------------
__global__ void node_attn(const float* __restrict__ h, const float* __restrict__ a_s,
                          const float* __restrict__ a_d, float* __restrict__ as_n,
                          float* __restrict__ ad_n, int N) {
    int i = blockIdx.x * blockDim.x + threadIdx.x;
    if (i >= N) return;
    const float* row = h + (long long)i * HID;
    float s = 0.f, t = 0.f;
    #pragma unroll
    for (int c = 0; c < HID; ++c) { s += row[c] * a_s[c]; t += row[c] * a_d[c]; }
    as_n[i] = s; ad_n[i] = t;
}

__global__ void edge_alpha(const int* __restrict__ src, const int* __restrict__ dst,
                           const float* __restrict__ edge_attr,
                           const float* __restrict__ loop_attr,
                           const float* __restrict__ we_ae,
                           const float* __restrict__ as_n, const float* __restrict__ ad_n,
                           float* __restrict__ exbuf, float* __restrict__ amax,
                           int E, int N) {
    int idx = blockIdx.x * blockDim.x + threadIdx.x;
    if (idx >= E + N) return;
    int s, d; const float* row;
    if (idx < E) { s = src[idx]; d = dst[idx]; row = edge_attr + (long long)idx * HID; }
    else         { s = d = idx - E;            row = loop_attr + (long long)s * HID; }
    float ae = 0.f;
    #pragma unroll
    for (int c = 0; c < HID; ++c) ae += row[c] * we_ae[c];
    float al = as_n[s] + ad_n[d] + ae;
    al = (al > 0.f) ? al : NEG_SLOPE * al;
    exbuf[idx] = al;
    atomicMaxF(&amax[d], al);
}

__global__ void edge_exp(const int* __restrict__ dst, float* __restrict__ exbuf,
                         const float* __restrict__ amax, float* __restrict__ denom,
                         int E, int N) {
    int idx = blockIdx.x * blockDim.x + threadIdx.x;
    if (idx >= E + N) return;
    int d = (idx < E) ? dst[idx] : (idx - E);
    float ex = expf(exbuf[idx] - amax[d]);
    exbuf[idx] = ex;
    atomicAdd(&denom[d], ex);
}

__global__ void edge_aggregate(const int* __restrict__ src, const int* __restrict__ dst,
                               const float* __restrict__ h, const float* __restrict__ exbuf,
                               const float* __restrict__ denom, float* __restrict__ accum,
                               int E, int N) {
    long long t = (long long)blockIdx.x * blockDim.x + threadIdx.x;
    if (t >= (long long)(E + N) * 8) return;
    int idx = (int)(t >> 3);
    int c0  = (int)(t & 7) * 8;
    int s, d;
    if (idx < E) { s = src[idx]; d = dst[idx]; } else { s = d = idx - E; }
    float coef = exbuf[idx] / denom[d];
    const float* hr = h + (long long)s * HID;
    float* acc = accum + (long long)d * HID;
    #pragma unroll
    for (int c = 0; c < 8; ++c) atomicAdd(&acc[c0 + c], hr[c0 + c] * coef);
}

__global__ void node_bias_act(const float* __restrict__ accum, const float* __restrict__ b,
                              float* __restrict__ out, int N, int mode) {
    long long t = (long long)blockIdx.x * blockDim.x + threadIdx.x;
    if (t >= (long long)N * HID) return;
    int c = (int)(t & 63);
    float v = accum[t] + b[c];
    if (mode == 1) v = (v > 0.f) ? v : (expf(v) - 1.f);
    out[t] = v;
}

// ---------------------------------------------------------------------------
// Fused edge MLP: e = relu(ea @ Wm1 + bm1) @ Wm2 + bm2.
// B1/B2 packs staged in LDS per block (no VGPR spills). GEMM1's D goes
// straight into bf16 A-fragment layout in wave-private LDS so GEMM2's A load
// is one contiguous 32B LDS read. Grid-stride loop over 16-edge tiles;
// final stores are nontemporal.
// ---------------------------------------------------------------------------
__global__ __launch_bounds__(256) void edge_mlp(const float* __restrict__ ea,
                                                const __bf16* __restrict__ B1p,
                                                const float* __restrict__ bm1,
                                                const __bf16* __restrict__ B2p,
                                                const float* __restrict__ bm2,
                                                float* __restrict__ eout,
                                                long long E) {
    __shared__ __bf16 B1s[2 * 4 * 32 * 16];          // 8 KB
    __shared__ __bf16 B2s[2 * 4 * 32 * 16];          // 8 KB
    __shared__ __bf16 midp[8][2 * 32 * 16];          // 16 KB (2 KB per wave)
    for (int t = threadIdx.x; t < 512; t += 256) {
        ((int4*)B1s)[t] = ((const int4*)B1p)[t];
        ((int4*)B2s)[t] = ((const int4*)B2p)[t];
    }
    __syncthreads();

    int lane = threadIdx.x & 31;
    int wave = threadIdx.x >> 5;
    __bf16* m = midp[wave];

    float bias1[4], bias2[4];
    #pragma unroll
    for (int nt = 0; nt < 4; ++nt) {
        bias1[nt] = bm1[nt * 16 + (lane & 15)];
        bias2[nt] = bm2[nt * 16 + (lane & 15)];
    }
    int mb = (lane < 16) ? 0 : 8;
    int hi = ((lane & 15) >= 8) ? 16 : 0;

    long long ntiles = (E + 15) >> 4;
    long long stride = (long long)gridDim.x * 8;
    for (long long tile = (long long)blockIdx.x * 8 + wave; tile < ntiles;
         tile += stride) {
        long long row0 = tile << 4;
        long long nxt = tile + stride;
        if (nxt < ntiles)
            __builtin_prefetch(ea + ((nxt << 4) + (lane & 15)) * HID, 0, 1);

        v16bf A0 = load_A_f32(ea, HID, row0, 0,  lane, E);
        v16bf A1 = load_A_f32(ea, HID, row0, 32, lane, E);

        // GEMM1 + bias + ReLU -> LDS in bf16 A-layout
        #pragma unroll
        for (int nt = 0; nt < 4; ++nt) {
            v8f acc = {};
            acc = wmma_bf16(A0, frag_B(B1s, 0, nt, lane), acc);
            acc = wmma_bf16(A1, frag_B(B1s, 1, nt, lane), acc);
            int kt = nt >> 1;
            int i  = ((nt & 1) ? 8 : 0) + (lane & 7);
            #pragma unroll
            for (int r = 0; r < 8; ++r) {
                float v = acc[r] + bias1[nt];
                v = (v > 0.f) ? v : 0.f;
                m[(kt * 32 + mb + r + hi) * 16 + i] = (__bf16)v;
            }
        }
        __builtin_amdgcn_wave_barrier();   // LDS in-order per wave; fence compiler

        v16bf C0 = *(const v16bf*)(m + (0 * 32 + lane) * 16);
        v16bf C1 = *(const v16bf*)(m + (1 * 32 + lane) * 16);

        // GEMM2 + bias -> global (nontemporal: written once, never re-read)
        #pragma unroll
        for (int nt = 0; nt < 4; ++nt) {
            v8f acc = {};
            acc = wmma_bf16(C0, frag_B(B2s, 0, nt, lane), acc);
            acc = wmma_bf16(C1, frag_B(B2s, 1, nt, lane), acc);
            int n = nt * 16 + (lane & 15);
            #pragma unroll
            for (int r = 0; r < 8; ++r) {
                long long row = row0 + mb + r;
                if (row < E)
                    __builtin_nontemporal_store(acc[r] + bias2[nt],
                                                &eout[row * HID + n]);
            }
        }
        __builtin_amdgcn_wave_barrier();   // before next iter overwrites mid
    }
}

// ---------------------------------------------------------------------------
// Host-side launcher
// ---------------------------------------------------------------------------
static inline int cdiv(long long a, long long b) { return (int)((a + b - 1) / b); }

extern "C" void kernel_launch(void* const* d_in, const int* in_sizes, int n_in,
                              void* d_out, int out_size, void* d_ws, size_t ws_size,
                              hipStream_t stream) {
    const float* x         = (const float*)d_in[0];
    const int*   ei        = (const int*)  d_in[1];
    const float* edge_attr = (const float*)d_in[2];
    const int*   batch     = (const int*)  d_in[3];
    const float* W1  = (const float*)d_in[4];
    const float* as1 = (const float*)d_in[5];
    const float* ad1 = (const float*)d_in[6];
    const float* We1 = (const float*)d_in[7];
    const float* ae1 = (const float*)d_in[8];
    const float* b1  = (const float*)d_in[9];
    const float* W2  = (const float*)d_in[10];
    const float* as2 = (const float*)d_in[11];
    const float* ad2 = (const float*)d_in[12];
    const float* We2 = (const float*)d_in[13];
    const float* ae2 = (const float*)d_in[14];
    const float* b2  = (const float*)d_in[15];
    const float* Wm1 = (const float*)d_in[16];
    const float* bm1 = (const float*)d_in[17];
    const float* Wm2 = (const float*)d_in[18];
    const float* bm2 = (const float*)d_in[19];

    const int IN = 128;
    const long long N = in_sizes[0] / IN;
    const long long E = in_sizes[2] / HID;
    const int Ei = (int)E, Ni = (int)N;
    const int E2 = Ei + Ni;
    const int* src = ei;
    const int* dst = ei + Ei;

    // ---- workspace carve: bf16 packed weights first (keeps 32B alignment) --
    char* wsb = (char*)d_ws;
    __bf16* W1p  = (__bf16*)wsb; wsb += (size_t)IN  * HID * 2;  // 16 KB
    __bf16* W2p  = (__bf16*)wsb; wsb += (size_t)HID * HID * 2;  //  8 KB
    __bf16* Wm1p = (__bf16*)wsb; wsb += (size_t)HID * HID * 2;
    __bf16* Wm2p = (__bf16*)wsb; wsb += (size_t)HID * HID * 2;
    float* ws = (float*)wsb;
    float* deg       = ws;  ws += N;
    float* loop_attr = ws;  ws += N * HID;
    float* h         = ws;  ws += N * HID;
    float* hact      = ws;  ws += N * HID;
    float* accum     = ws;  ws += N * HID;
    float* as_n      = ws;  ws += N;
    float* ad_n      = ws;  ws += N;
    float* amax      = ws;  ws += N;
    float* denom     = ws;  ws += N;
    float* exbuf     = ws;  ws += (long long)E2;
    float* we_ae1    = ws;  ws += HID;
    float* we_ae2    = ws;  ws += HID;

    float* out_h = (float*)d_out;            // [N x 64]
    float* out_e = out_h + N * HID;          // [E x 64]
    int*   out_b = (int*)(out_e + E * HID);  // [N]

    const int T = 256;

    // ---- init + packing + small precomputes -------------------------------
    fill_f32<<<cdiv(N, T), T, 0, stream>>>(deg, 0.f, N);
    fill_f32<<<cdiv(N * HID, T), T, 0, stream>>>(loop_attr, 0.f, N * HID);
    fill_f32<<<cdiv(N, T), T, 0, stream>>>(amax, -INFINITY, N);
    fill_f32<<<cdiv(N, T), T, 0, stream>>>(denom, 0.f, N);
    fill_f32<<<cdiv(N * HID, T), T, 0, stream>>>(accum, 0.f, N * HID);
    pack_weight<<<cdiv(IN * HID, T), T, 0, stream>>>(W1, W1p, IN);
    pack_weight<<<cdiv(HID * HID, T), T, 0, stream>>>(W2, W2p, HID);
    pack_weight<<<cdiv(HID * HID, T), T, 0, stream>>>(Wm1, Wm1p, HID);
    pack_weight<<<cdiv(HID * HID, T), T, 0, stream>>>(Wm2, Wm2p, HID);
    matvec64<<<1, 64, 0, stream>>>(We1, ae1, we_ae1);
    matvec64<<<1, 64, 0, stream>>>(We2, ae2, we_ae2);

    edge_deg_loop<<<cdiv((long long)Ei * 8, T), T, 0, stream>>>(dst, edge_attr, deg, loop_attr, Ei);
    finalize_loop<<<cdiv(N * HID, T), T, 0, stream>>>(loop_attr, deg, Ni);

    // ---- GAT layer 1 ------------------------------------------------------
    {
        long long ntiles = (N + 15) >> 4;
        int grid = cdiv(ntiles, 8 * 4);              // ~4 tiles per wave
        gemm_proj_t<4><<<grid, 256, 0, stream>>>(x, W1p, h, N);
    }
    node_attn<<<cdiv(N, T), T, 0, stream>>>(h, as1, ad1, as_n, ad_n, Ni);
    edge_alpha<<<cdiv(E2, T), T, 0, stream>>>(src, dst, edge_attr, loop_attr, we_ae1,
                                              as_n, ad_n, exbuf, amax, Ei, Ni);
    edge_exp<<<cdiv(E2, T), T, 0, stream>>>(dst, exbuf, amax, denom, Ei, Ni);
    edge_aggregate<<<cdiv((long long)E2 * 8, T), T, 0, stream>>>(src, dst, h, exbuf, denom,
                                                                 accum, Ei, Ni);
    node_bias_act<<<cdiv(N * HID, T), T, 0, stream>>>(accum, b1, hact, Ni, /*ELU*/1);

    // ---- re-init softmax state -------------------------------------------
    fill_f32<<<cdiv(N, T), T, 0, stream>>>(amax, -INFINITY, N);
    fill_f32<<<cdiv(N, T), T, 0, stream>>>(denom, 0.f, N);
    fill_f32<<<cdiv(N * HID, T), T, 0, stream>>>(accum, 0.f, N * HID);

    // ---- GAT layer 2 ------------------------------------------------------
    {
        long long ntiles = (N + 15) >> 4;
        int grid = cdiv(ntiles, 8 * 4);
        gemm_proj_t<2><<<grid, 256, 0, stream>>>(hact, W2p, h, N);
    }
    node_attn<<<cdiv(N, T), T, 0, stream>>>(h, as2, ad2, as_n, ad_n, Ni);
    edge_alpha<<<cdiv(E2, T), T, 0, stream>>>(src, dst, edge_attr, loop_attr, we_ae2,
                                              as_n, ad_n, exbuf, amax, Ei, Ni);
    edge_exp<<<cdiv(E2, T), T, 0, stream>>>(dst, exbuf, amax, denom, Ei, Ni);
    edge_aggregate<<<cdiv((long long)E2 * 8, T), T, 0, stream>>>(src, dst, h, exbuf, denom,
                                                                 accum, Ei, Ni);
    node_bias_act<<<cdiv(N * HID, T), T, 0, stream>>>(accum, b2, out_h, Ni, /*identity*/0);

    // ---- edge MLP (fused double WMMA GEMM) --------------------------------
    {
        long long ntiles = (E + 15) >> 4;
        int grid = cdiv(ntiles, 8 * 4);              // ~4 tiles per wave
        if (grid > 2048) grid = 2048;
        edge_mlp<<<grid, 256, 0, stream>>>(edge_attr, Wm1p, bm1, Wm2p, bm2, out_e, E);
    }

    // ---- pass-through batch ----------------------------------------------
    copy_i32<<<cdiv(N, T), T, 0, stream>>>(batch, out_b, Ni);
}
